// KAN_Convolutional_Layer_13529146982428
// MI455X (gfx1250) — matrix operand order repaired and checked by reading, not compiled, not run on hardware
//
#include <hip/hip_runtime.h>

// ---- problem constants (from reference) ----
#define C_IN     8
#define C_OUT    32
#define HH       64
#define WW       64
#define HWPIX    4096          // 64*64
#define NPIX     65536         // 16*64*64
#define NGRP     72            // C_IN * 9 taps
#define K_REAL   648           // NGRP * 9 features (silu + 8 bases)
#define K_PAD    672           // 21 * 32, WMMA-friendly
#define KNOT0   (-2.2f)
#define KNOTH    0.4f          // (1-(-1))/5

typedef __attribute__((ext_vector_type(16))) _Float16 v16h;
typedef __attribute__((ext_vector_type(8)))  _Float16 v8h;
typedef __attribute__((ext_vector_type(8)))  float    v8f;

union Frag16 { v16h v; v8h h[2]; };
union CFrag  { v8f  v; float4 f[2]; };

// ------------------------------------------------------------------
// Kernel 0: pack combined weight matrix Wq[o][col] (f16, K-major rows)
//   col = g*9 + j,  g = c*9 + k.  j==0 -> base_weight, j==1+s -> sw*ss.
// ------------------------------------------------------------------
__global__ void kan_pack_weights(const float* __restrict__ bw,
                                 const float* __restrict__ sw,
                                 const float* __restrict__ ss,
                                 _Float16* __restrict__ Wq) {
  int tid = blockIdx.x * 256 + threadIdx.x;
  if (tid >= C_OUT * K_PAD) return;
  int o = tid / K_PAD, col = tid - o * K_PAD;
  float v = 0.0f;
  if (col < K_REAL) {
    int g = col / 9, j = col - g * 9;
    int idx = o * NGRP + g;                       // (o, c, k) flat
    v = (j == 0) ? bw[idx] : sw[idx * 8 + (j - 1)] * ss[idx];
  }
  Wq[tid] = (_Float16)v;
}

// ------------------------------------------------------------------
// Kernel 1: feature build.  One thread per (pixel, group); g==72 zeros
// the K padding.  De Boor on a uniform knot vector (denoms = 0.4*k).
// Out-of-image taps use x=0 (reference pads BEFORE silu/b_splines).
// ------------------------------------------------------------------
__global__ void kan_features(const float* __restrict__ x,
                             _Float16* __restrict__ F,
                             int base_pixel, int rows) {
  int tid = blockIdx.x * 256 + threadIdx.x;
  if (tid >= rows * 73) return;
  int pl = tid / 73, g = tid - pl * 73;
  long fbase = (long)pl * K_PAD;

  if (g == NGRP) {                                // zero pad cols 648..671
    v8h z = {};
    *(v8h*)(F + fbase + 648) = z;
    *(v8h*)(F + fbase + 656) = z;
    *(v8h*)(F + fbase + 664) = z;
    return;
  }

  int c = g / 9, k = g - c * 9;
  int dh = k / 3 - 1, dw = k % 3 - 1;
  int n = base_pixel + pl;
  int b = n >> 12, h = (n >> 6) & 63, w = n & 63;
  int ih = h + dh, iw = w + dw;
  float xv = 0.0f;
  if ((unsigned)ih < (unsigned)HH && (unsigned)iw < (unsigned)WW)
    xv = x[((size_t)b * C_IN + c) * HWPIX + ih * WW + iw];

  float silu = xv / (1.0f + __expf(-xv));

  float bas[11];
#pragma unroll
  for (int i = 0; i < 11; ++i) {
    float ti = KNOT0 + KNOTH * i;
    bas[i] = (xv >= ti && xv < ti + KNOTH) ? 1.0f : 0.0f;
  }
#pragma unroll
  for (int kk = 1; kk <= 3; ++kk) {
    float inv = 1.0f / (KNOTH * kk);
#pragma unroll
    for (int i = 0; i < 11 - kk; ++i) {
      float ti = KNOT0 + KNOTH * i;
      bas[i] = (xv - ti) * inv * bas[i] +
               ((ti + KNOTH * (kk + 1)) - xv) * inv * bas[i + 1];
    }
  }

  _Float16* dst = F + fbase + g * 9;
  dst[0] = (_Float16)silu;
#pragma unroll
  for (int s = 0; s < 8; ++s) dst[1 + s] = (_Float16)bas[s];
}

// ------------------------------------------------------------------
// Kernel 2: WMMA GEMM.  256 threads = 8 waves, 256 rows per block.
// B matrix (42KB) staged in LDS once per block; each wave computes a
// 32(M) x 32(N) tile with 4 f32 accumulators over K=672, so every
// fragment load feeds two v_wmma_f32_16x16x32_f16.  Per K-step:
// 4 global b128 (A) + 4 ds b128 (B) + 4 WMMA.
// ------------------------------------------------------------------
__global__ void __launch_bounds__(256)
kan_wmma_gemm(const _Float16* __restrict__ F,
              const _Float16* __restrict__ Wq,
              float* __restrict__ out,
              int base_pixel) {
  __shared__ _Float16 Bs[C_OUT * K_PAD];          // 43008 B of 320KB/WGP

  // cooperative stage: 2688 x 16B chunks across 256 threads
  for (int i = threadIdx.x; i < (C_OUT * K_PAD) / 8; i += 256)
    ((v8h*)Bs)[i] = ((const v8h*)Wq)[i];
  __syncthreads();

  int lane = threadIdx.x & 31;
  int wave = threadIdx.x >> 5;
  long row0 = (long)blockIdx.x * 256 + wave * 32; // 32 rows per wave
  int nl = lane & 15;                             // M row / N col in tile
  int hi = lane >> 4;

  const _Float16* A0 = F + (row0 + nl) * K_PAD;        // M-tile 0
  const _Float16* A1 = F + (row0 + 16 + nl) * K_PAD;   // M-tile 1
  const _Float16* B0 = Bs + nl * K_PAD + hi * 16;      // N-half 0 (o = nl)
  const _Float16* B1 = Bs + (nl + 16) * K_PAD + hi * 16; // N-half 1

  v8f c00 = {}, c01 = {}, c10 = {}, c11 = {};
#pragma unroll 3
  for (int kb = 0; kb < K_PAD; kb += 32) {
    Frag16 a0, a1, b0, b1;
    // A fragment in contiguous k order (B matches): 16 K at kb + hi*16 for this lane's row
    a0.h[0] = *(const v8h*)(A0 + kb + hi * 16);
    a0.h[1] = *(const v8h*)(A0 + kb + hi * 16 + 8);
    a1.h[0] = *(const v8h*)(A1 + kb + hi * 16);
    a1.h[1] = *(const v8h*)(A1 + kb + hi * 16 + 8);
    // B fragment in contiguous k order (matches A): 16 K at kb + hi*16 for this lane's column
    b0.h[0] = *(const v8h*)(B0 + kb);
    b0.h[1] = *(const v8h*)(B0 + kb + 8);
    b1.h[0] = *(const v8h*)(B1 + kb);
    b1.h[1] = *(const v8h*)(B1 + kb + 8);
    c00 = __builtin_amdgcn_wmma_f32_16x16x32_f16(false, a0.v, false, b0.v,
                                                 (short)0, c00, false, false);
    c01 = __builtin_amdgcn_wmma_f32_16x16x32_f16(false, a0.v, false, b1.v,
                                                 (short)0, c01, false, false);
    c10 = __builtin_amdgcn_wmma_f32_16x16x32_f16(false, a1.v, false, b0.v,
                                                 (short)0, c10, false, false);
    c11 = __builtin_amdgcn_wmma_f32_16x16x32_f16(false, a1.v, false, b1.v,
                                                 (short)0, c11, false, false);
  }

  // C/D layout: VGPR r -> M = 8*hi + r, N = nl.  Each accumulator holds
  // 8 consecutive pixels for one channel -> two b128 stores each.
  // A 16-row tile never crosses a 4096-pixel batch image.
  long p0 = (long)base_pixel + row0 + 8 * hi;     // M-tile 0 pixel base
  long p1 = p0 + 16;                              // M-tile 1 pixel base
  float* o0 = out + (p0 >> 12) * (C_OUT * HWPIX) + (p0 & 4095);
  float* o1 = out + (p1 >> 12) * (C_OUT * HWPIX) + (p1 & 4095);
  CFrag r00, r01, r10, r11;
  r00.v = c00; r01.v = c01; r10.v = c10; r11.v = c11;
  float* q = o0 + (size_t)nl * HWPIX;
  *(float4*)(q)     = r00.f[0];
  *(float4*)(q + 4) = r00.f[1];
  q = o0 + (size_t)(nl + 16) * HWPIX;
  *(float4*)(q)     = r01.f[0];
  *(float4*)(q + 4) = r01.f[1];
  q = o1 + (size_t)nl * HWPIX;
  *(float4*)(q)     = r10.f[0];
  *(float4*)(q + 4) = r10.f[1];
  q = o1 + (size_t)(nl + 16) * HWPIX;
  *(float4*)(q)     = r11.f[0];
  *(float4*)(q + 4) = r11.f[1];
}

// ------------------------------------------------------------------
extern "C" void kernel_launch(void* const* d_in, const int* in_sizes, int n_in,
                              void* d_out, int out_size, void* d_ws, size_t ws_size,
                              hipStream_t stream) {
  const float* x  = (const float*)d_in[0];
  const float* bw = (const float*)d_in[1];
  const float* sw = (const float*)d_in[2];
  const float* ss = (const float*)d_in[3];
  float* out = (float*)d_out;

  // ws layout: [Wq: 32*672 f16 = 43008 B][64KB align][F: chunk*672 f16]
  _Float16* Wq = (_Float16*)d_ws;
  _Float16* F  = (_Float16*)((char*)d_ws + 65536);

  long avail = (long)ws_size - 65536;
  long chunk = (avail > 0) ? avail / (K_PAD * 2) : 0;   // pixels that fit
  chunk &= ~255L;                                       // multiple of 256 rows
  if (chunk < 256)   chunk = 256;
  if (chunk > NPIX)  chunk = NPIX;

  kan_pack_weights<<<(C_OUT * K_PAD + 255) / 256, 256, 0, stream>>>(bw, sw, ss, Wq);

  for (long base = 0; base < NPIX; base += chunk) {
    long rows = NPIX - base;
    if (rows > chunk) rows = chunk;
    long nthr = rows * 73;
    kan_features<<<(int)((nthr + 255) / 256), 256, 0, stream>>>(x, F, (int)base, (int)rows);
    kan_wmma_gemm<<<(int)(rows / 256), 256, 0, stream>>>(F, Wq, out, (int)base);
  }
}
